// GCNClassifier_82119774699583
// MI455X (gfx1250) — matrix operand order repaired and checked by reading, not compiled, not run on hardware
//
#include <hip/hip_runtime.h>
#include <math.h>

// MI455X / gfx1250: wave32, WMMA 16x16x32 f16->f32.
typedef __attribute__((ext_vector_type(16))) _Float16 v16h;
typedef __attribute__((ext_vector_type(8)))  float    v8f;
typedef __attribute__((ext_vector_type(4)))  _Float16 v4h;

#define HID 64
#define MAX_K 256   // largest GEMM K (layer 1); LDS stage = K*64 halves <= 32KB

// ---- degree / normalization -------------------------------------------------
__global__ void deg_init_kernel(float* deg, int N) {
  int i = blockIdx.x * blockDim.x + threadIdx.x;
  if (i < N) deg[i] = 1.0f;  // self-loop contributes 1 to every in-degree
}

__global__ void deg_count_kernel(const long long* __restrict__ dst, float* deg, int E) {
  int e = blockIdx.x * blockDim.x + threadIdx.x;
  if (e < E) atomicAdd(&deg[(int)dst[e]], 1.0f);
}

__global__ void dinv_kernel(const float* __restrict__ deg, float* __restrict__ dinv, int N) {
  int i = blockIdx.x * blockDim.x + threadIdx.x;
  if (i < N) dinv[i] = rsqrtf(deg[i]);  // deg >= 1 always (self-loops)
}

// ---- f32 -> f16 conversion (4-wide) ----------------------------------------
__global__ void cvt_f16_kernel(const float* __restrict__ x, _Float16* __restrict__ xh, int n4) {
  int i = blockIdx.x * blockDim.x + threadIdx.x;
  if (i >= n4) return;
  float4 v = ((const float4*)x)[i];
  v4h o = { (_Float16)v.x, (_Float16)v.y, (_Float16)v.z, (_Float16)v.w };
  ((v4h*)xh)[i] = o;
}

// ---- pack W[K,64] into per-lane WMMA B layout ------------------------------
// B tile (32x16 f16, 8 VGPRs), contiguous k order chosen here (same as A): lanes 0-15 hold K=0..15 of column n, lanes
// 16-31 hold K=16..31.  Packed so the GEMM reads 16 contiguous halves per
// lane per (ktile,ntile): Wp[((t*4+u)*32 + lane)*16 + q]
__global__ void pack_b_kernel(const float* __restrict__ W, _Float16* __restrict__ Wp, int K) {
  int idx = blockIdx.x * blockDim.x + threadIdx.x;
  int total = (K >> 5) * 4 * 32 * 16;
  if (idx >= total) return;
  int q = idx & 15;
  int l = (idx >> 4) & 31;
  int u = (idx >> 9) & 3;
  int t = idx >> 11;
  int k = t * 32 + (l >> 4) * 16 + q;
  int n = u * 16 + (l & 15);
  Wp[idx] = (_Float16)W[k * HID + n];
}

// ---- WMMA GEMM: C[M,64] = A[M,K] (f16, row major) x Wp (packed) ------------
// One wave computes a 16x64 stripe: 4 accumulators of 16x16, K-loop step 32.
// B is staged once per block into LDS (shared by all 8 waves); A is software-
// pipelined one k-tile ahead so global loads overlap the WMMA issue.
__global__ void __launch_bounds__(256)
gemm_wmma_kernel(const _Float16* __restrict__ A, const _Float16* __restrict__ Bp,
                 float* __restrict__ C, int M, int K) {
  __shared__ _Float16 Bs[MAX_K * HID];   // 32KB max; WGP has 320KB LDS

  // cooperative stage of packed W into LDS (16B per thread per step)
  int nhalves = K * HID;
  for (int idx = threadIdx.x * 8; idx < nhalves; idx += 256 * 8)
    *(float4*)(Bs + idx) = *(const float4*)(Bp + idx);
  __syncthreads();

  int wave = blockIdx.x * (blockDim.x >> 5) + (threadIdx.x >> 5);
  int lane = threadIdx.x & 31;
  if (wave >= (M >> 4)) return;           // uniform per wave; EXEC stays all-1s
  int hsel = lane >> 4;                    // half-wave select
  int row  = (wave << 4) + (lane & 15);    // A: M = lane&15 in both half-waves

  // A 16x32 f16, contiguous k order chosen here: lanes0-15 K=kk..kk+15, lanes16-31 K=kk+16..kk+31
  // (matches packed B).  Two contiguous b128 loads per lane.
  const _Float16* arow = A + (size_t)row * K + 16 * hsel;

  v8f acc[4] = {v8f{}, v8f{}, v8f{}, v8f{}};
  int kt = K >> 5;

  v16h a;
  *(float4*)&a       = *(const float4*)(arow);
  *((float4*)&a + 1) = *(const float4*)(arow + 8);

  for (int t = 0; t < kt; ++t) {
    v16h an = a;
    if (t + 1 < kt) {                      // prefetch next A k-tile
      const _Float16* ap = arow + ((t + 1) << 5);
      *(float4*)&an       = *(const float4*)(ap);
      *((float4*)&an + 1) = *(const float4*)(ap + 8);
    }
#pragma unroll
    for (int u = 0; u < 4; ++u) {
      const _Float16* bp = Bs + ((((t << 2) | u) * 32 + lane) << 4);
      v16h b;
      *(float4*)&b       = *(const float4*)(bp);      // ds_load_b128
      *((float4*)&b + 1) = *(const float4*)(bp + 8);  // ds_load_b128
      acc[u] = __builtin_amdgcn_wmma_f32_16x16x32_f16(
          false, a, false, b, (short)0, acc[u], false, false);
    }
    a = an;
  }

  // C/D 16x16 f32 layout: lane<16 -> M=j, lane>=16 -> M=j+8; N = lane&15.
  int col = lane & 15;
  int r0  = (wave << 4) + (hsel << 3);
#pragma unroll
  for (int u = 0; u < 4; ++u)
#pragma unroll
    for (int j = 0; j < 8; ++j)
      C[(size_t)(r0 + j) * HID + u * 16 + col] = acc[u][j];
}

// ---- zero scratch ----------------------------------------------------------
__global__ void zero_kernel(float* p, int n) {
  int i = blockIdx.x * blockDim.x + threadIdx.x;
  if (i < n) p[i] = 0.0f;
}

// ---- edge aggregation: agg[d] += h[s] * dinv[s]*dinv[d] --------------------
// 16 threads per edge, float4 per thread; h/agg are L2-resident (12.8MB each).
__global__ void aggregate_kernel(const float* __restrict__ h,
                                 const long long* __restrict__ src,
                                 const long long* __restrict__ dst,
                                 const float* __restrict__ dinv,
                                 float* agg, int E) {
  int gid = blockIdx.x * blockDim.x + threadIdx.x;
  int e = gid >> 4, c = gid & 15;
  if (e >= E) return;
  int s = (int)src[e], d = (int)dst[e];
  float nm = dinv[s] * dinv[d];
  float4 v = ((const float4*)h)[(size_t)s * 16 + c];
  float* o = agg + (size_t)d * HID + c * 4;
  atomicAdd(o + 0, v.x * nm);   // no-return global_atomic_add_f32
  atomicAdd(o + 1, v.y * nm);
  atomicAdd(o + 2, v.z * nm);
  atomicAdd(o + 3, v.w * nm);
}

// ---- self-loop + bias + ReLU; writes f32 in place, optional f16 copy -------
__global__ void finalize_kernel(float* agg, const float* __restrict__ h,
                                const float* __restrict__ dinv,
                                const float* __restrict__ b,
                                _Float16* outh, int N) {
  int gid = blockIdx.x * blockDim.x + threadIdx.x;
  if (gid >= N * HID) return;
  int i = gid >> 6, k = gid & 63;
  float di = dinv[i];
  float v = agg[gid] + h[gid] * di * di + b[k];
  v = fmaxf(v, 0.0f);
  agg[gid] = v;
  if (outh) outh[gid] = (_Float16)v;
}

// ---- classifier: sigmoid(h2 @ Wc + bc) -------------------------------------
__global__ void classify_kernel(const float* __restrict__ h, const float* __restrict__ Wc,
                                const float* __restrict__ bc, float* __restrict__ out, int N) {
  int i = blockIdx.x * blockDim.x + threadIdx.x;
  if (i >= N) return;
  const float4* hv = (const float4*)(h + (size_t)i * HID);
  const float4* wv = (const float4*)Wc;
  float s = 0.0f;
#pragma unroll
  for (int k = 0; k < 16; ++k) {
    float4 a = hv[k], w = wv[k];
    s += a.x * w.x + a.y * w.y + a.z * w.z + a.w * w.w;
  }
  s += bc[0];
  out[i] = 1.0f / (1.0f + expf(-s));
}

extern "C" void kernel_launch(void* const* d_in, const int* in_sizes, int n_in,
                              void* d_out, int out_size, void* d_ws, size_t ws_size,
                              hipStream_t stream) {
  (void)n_in; (void)out_size; (void)ws_size;
  const float*     x  = (const float*)d_in[0];
  const long long* ei = (const long long*)d_in[1];   // int64 edge_index [2,E]
  const float*     W1 = (const float*)d_in[2];
  const float*     b1 = (const float*)d_in[3];
  const float*     W2 = (const float*)d_in[4];
  const float*     b2 = (const float*)d_in[5];
  const float*     Wc = (const float*)d_in[6];
  const float*     bc = (const float*)d_in[7];

  const int IN = 256;
  const int N = in_sizes[0] / IN;    // 50000 (multiple of 16)
  const int E = in_sizes[1] / 2;     // 800000
  const long long* src = ei;
  const long long* dst = ei + E;

  // workspace carve-out (~58 MB total)
  char* w = (char*)d_ws;
  auto alloc = [&](size_t bytes) -> char* {
    char* p = w; w += (bytes + 255) & ~(size_t)255; return p;
  };
  float*    deg  = (float*)alloc((size_t)N * 4);
  float*    dinv = (float*)alloc((size_t)N * 4);
  _Float16* xh   = (_Float16*)alloc((size_t)N * IN * 2);
  _Float16* wp1  = (_Float16*)alloc((size_t)(IN >> 5) * 2048 * 2);
  _Float16* wp2  = (_Float16*)alloc((size_t)(HID >> 5) * 2048 * 2);
  float*    hbuf = (float*)alloc((size_t)N * HID * 4);
  float*    agg  = (float*)alloc((size_t)N * HID * 4);
  _Float16* h1h  = (_Float16*)alloc((size_t)N * HID * 2);

  const int T = 256;
  auto cdiv = [](long long a, long long b) { return (int)((a + b - 1) / b); };

  // normalization
  deg_init_kernel<<<cdiv(N, T), T, 0, stream>>>(deg, N);
  deg_count_kernel<<<cdiv(E, T), T, 0, stream>>>(dst, deg, E);
  dinv_kernel<<<cdiv(N, T), T, 0, stream>>>(deg, dinv, N);

  // operand prep
  cvt_f16_kernel<<<cdiv((long long)N * IN / 4, T), T, 0, stream>>>(x, xh, N * IN / 4);
  pack_b_kernel<<<cdiv((IN >> 5) * 2048, T), T, 0, stream>>>(W1, wp1, IN);
  pack_b_kernel<<<cdiv((HID >> 5) * 2048, T), T, 0, stream>>>(W2, wp2, HID);

  // ---- layer 1: relu(norm_agg(x @ W1) + b1) ----
  gemm_wmma_kernel<<<cdiv(N >> 4, 8), 256, 0, stream>>>(xh, wp1, hbuf, N, IN);
  zero_kernel<<<cdiv((long long)N * HID, T), T, 0, stream>>>(agg, N * HID);
  aggregate_kernel<<<cdiv((long long)E * 16, T), T, 0, stream>>>(hbuf, src, dst, dinv, agg, E);
  finalize_kernel<<<cdiv((long long)N * HID, T), T, 0, stream>>>(agg, hbuf, dinv, b1, h1h, N);

  // ---- layer 2: relu(norm_agg(h1 @ W2) + b2) ----
  gemm_wmma_kernel<<<cdiv(N >> 4, 8), 256, 0, stream>>>(h1h, wp2, hbuf, N, HID);
  zero_kernel<<<cdiv((long long)N * HID, T), T, 0, stream>>>(agg, N * HID);
  aggregate_kernel<<<cdiv((long long)E * 16, T), T, 0, stream>>>(hbuf, src, dst, dinv, agg, E);
  finalize_kernel<<<cdiv((long long)N * HID, T), T, 0, stream>>>(agg, hbuf, dinv, b2, (_Float16*)nullptr, N);

  // ---- classifier ----
  classify_kernel<<<cdiv(N, T), T, 0, stream>>>(agg, Wc, bc, (float*)d_out, N);
}
